// DeBERTaV3CrossAttention_10153302688059
// MI455X (gfx1250) — compile-verified, matrix operand order checked by reading, not run on hardware
//
#include <hip/hip_runtime.h>
#include <math.h>

#define B_  128
#define S_  1024
#define H_  768
#define NH_ 8
#define DH_ 96
#define NF_ 8
#define FUSED_ 2312
#define CH_ 32   // sequence chunk (rows per flash step)

typedef __attribute__((ext_vector_type(16))) __bf16 v16bf;
typedef __attribute__((ext_vector_type(8)))  float  v8f;

// f32 -> bf16 round-to-nearest-even (bit-level, no __bf16 arithmetic needed)
__device__ __forceinline__ unsigned short f2bf(float f) {
  unsigned int u = __float_as_uint(f);
  u += 0x7FFFu + ((u >> 16) & 1u);
  return (unsigned short)(u >> 16);
}

// Assemble a 16-element bf16 fragment from LDS: elements [0..7] at p, [8..15] at p+16
// (CDNA5 16-bit A/B layout: per lane, K runs of 8 at Kbase and Kbase+16)
__device__ __forceinline__ v16bf frag16(const unsigned short* p) {
  union { uint4 u[2]; v16bf v; } t;
  t.u[0] = *(const uint4*)p;
  t.u[1] = *(const uint4*)(p + 16);
  return t.v;
}

// ---------------- Kernel A: masked mean over aspect span -> asp_query [B,H] ----
__global__ void __launch_bounds__(256)
asp_mean_kernel(const float* __restrict__ x, const int* __restrict__ sep1,
                const int* __restrict__ sep2, float* __restrict__ aspq) {
  int b = blockIdx.x;
  int s1 = sep1[b], s2 = sep2[b];
  bool cond = (s2 > s1 + 1);
  int lo = cond ? s1 + 1 : 0;
  int hi = cond ? s2 : 1;
  float inv = 1.0f / (float)(hi - lo);
  for (int j = threadIdx.x; j < H_; j += blockDim.x) {
    float acc = 0.f;
    const float* p = x + ((size_t)b * S_ + lo) * H_ + j;
    for (int s = lo; s < hi; ++s) { acc += *p; p += H_; }
    aspq[b * H_ + j] = acc * inv;
  }
}

// ---------------- Kernel B: q = aspq@Wq^T+bq ; T[b,h,:] = Wk_h^T q_bh (bf16, padded 16 heads);
//                  sbias[b,h] = q_bh . bk_h -------------------------------------
__global__ void __launch_bounds__(256)
qT_kernel(const float* __restrict__ aspq, const float* __restrict__ w,
          const float* __restrict__ bias, unsigned short* __restrict__ Tbf,
          float* __restrict__ sbias) {
  __shared__ float aq[H_];
  __shared__ float q[H_];
  int b = blockIdx.x;
  for (int j = threadIdx.x; j < H_; j += blockDim.x) aq[j] = aspq[b * H_ + j];
  __syncthreads();
  for (int i = threadIdx.x; i < H_; i += blockDim.x) {
    const float* wr = w + (size_t)i * H_;          // Wq row i
    float acc = bias[i];
    for (int j = 0; j < H_; ++j) acc += wr[j] * aq[j];
    q[i] = acc;
  }
  __syncthreads();
  for (int idx = threadIdx.x; idx < 16 * H_; idx += blockDim.x) {
    int h = idx / H_, j = idx - h * H_;
    float acc = 0.f;
    if (h < NH_) {
      for (int d = 0; d < DH_; ++d)
        acc += w[((size_t)(H_ + h * DH_ + d)) * H_ + j] * q[h * DH_ + d];
    }
    Tbf[(size_t)b * 16 * H_ + idx] = f2bf(acc);
  }
  if (threadIdx.x < 16) {
    int h = threadIdx.x;
    float acc = 0.f;
    if (h < NH_)
      for (int d = 0; d < DH_; ++d) acc += q[h * DH_ + d] * bias[H_ + h * DH_ + d];
    sbias[b * 16 + h] = acc;
  }
}

// ---------------- Kernel C: flash-style masked attention over text span --------
// Single pass over x[b, 0..sep1). Per 32-row chunk:
//   scores P = T(16x768,bf16) x chunk^T : K split over 8 waves (v_wmma bf16),
//     partials reduced in LDS by the softmax-stats phase,
//   online softmax stats; u += exp(P) x chunk (WMMA), rescaled each step.
// Writes ubar[b,16,768] = u / l.
__global__ void __launch_bounds__(256)
attn_kernel(const float* __restrict__ x, const int* __restrict__ sep1,
            const unsigned short* __restrict__ Tbf, const float* __restrict__ sbias,
            float* __restrict__ ubar) {
  extern __shared__ char smem[];
  unsigned short* T_s  = (unsigned short*)smem;        // 16*768 bf16   (24576 B)
  unsigned short* xrow = T_s + 16 * H_;                // 32*768 bf16   (49152 B)
  unsigned short* xT   = xrow + CH_ * H_;              // 768*32 bf16   (49152 B)
  float* pp            = (float*)(xT + H_ * CH_);      // 4*16*32 f32 partials (8192 B)
  unsigned short* pexp = (unsigned short*)(pp + 4 * 16 * CH_); // 16*32 bf16 (1024 B)
  float* mrow = (float*)(pexp + 16 * CH_);
  float* lrow = mrow + 16;
  float* frow = lrow + 16;
  float* sb   = frow + 16;

  int b = blockIdx.x;
  int tid = threadIdx.x, lane = tid & 31, wave = tid >> 5;
  int s1 = sep1[b];
  bool tcond = (s1 > 1);
  int nend = tcond ? s1 : 1;
  int nchunks = (nend + CH_ - 1) / CH_;

  for (int i = tid; i < 16 * H_; i += 256) T_s[i] = Tbf[(size_t)b * 16 * H_ + i];
  if (tid < 16) { sb[tid] = sbias[b * 16 + tid]; mrow[tid] = -1e30f; lrow[tid] = 0.f; }
  __syncthreads();

  const int mb = (lane < 16) ? 0 : 8;   // C/D row base for this lane-half
  const int nn = lane & 15;             // C/D column (N) within tile
  const int kb = (lane < 16) ? 0 : 8;   // A/B K base for this lane-half
  const int stile = wave >> 2;          // score tile (positions 0-15 / 16-31)
  const int ks    = wave & 3;           // score K-quarter (192 each)

  v8f acc[6];
#pragma unroll
  for (int t = 0; t < 6; ++t) acc[t] = {};
  const int j0 = wave * 96;             // this wave owns 96 output columns of u
  const float scale = 0.10206207261596575f;  // 1/sqrt(96)

  for (int c = 0; c < nchunks; ++c) {
    int c0 = c * CH_;
    // ---- stage chunk: global f32 -> bf16 LDS, row-major + transposed copies
    const float* xg = x + ((size_t)b * S_ + c0) * H_;
    for (int i = tid; i < CH_ * H_; i += 256) {
      int s = i / H_, j = i - s * H_;
      unsigned short v = f2bf(xg[i]);
      xrow[i] = v;
      xT[j * CH_ + s] = v;
    }
    if (c + 1 < nchunks)  // hint next chunk into caches (global_prefetch_b8)
      __builtin_prefetch(x + ((size_t)b * S_ + c0 + CH_) * H_ + tid * 96, 0, 1);
    __syncthreads();

    // ---- scores: all 8 waves, each does a (tile, K-quarter) partial of P
    {
      v8f sc = {};
      const unsigned short* ar = T_s + (size_t)nn * H_ + kb;
      const unsigned short* br = xrow + (size_t)(stile * 16 + nn) * H_ + kb;
#pragma unroll
      for (int kt = ks * 6; kt < ks * 6 + 6; ++kt) {
        v16bf a  = frag16(ar + kt * 32);
        v16bf bb = frag16(br + kt * 32);
        sc = __builtin_amdgcn_wmma_f32_16x16x32_bf16(false, a, false, bb,
                                                     (short)0, sc, false, false);
      }
      int sl = stile * 16 + nn;
#pragma unroll
      for (int r = 0; r < 8; ++r)
        pp[(ks * 16 + (r + mb)) * CH_ + sl] = sc[r];
    }
    __syncthreads();

    // ---- reduce partials + online softmax stats, one thread per head row
    if (tid < 16) {
      int m = tid;
      float p[CH_];
      float rm = -1e30f;
      for (int i = 0; i < CH_; ++i) {
        float s = pp[m * CH_ + i] + pp[(16 + m) * CH_ + i] +
                  pp[(32 + m) * CH_ + i] + pp[(48 + m) * CH_ + i];
        int pos = c0 + i;
        bool valid = tcond ? (pos >= 1 && pos < s1) : (pos == 0);
        p[i] = valid ? ((s + sb[m]) * scale) : -1e30f;   // bias inside the scale
        rm = fmaxf(rm, p[i]);
      }
      float mo = mrow[m];
      float mn = fmaxf(mo, rm);
      float f  = __expf(mo - mn);
      float ls = lrow[m] * f;
      for (int i = 0; i < CH_; ++i) {
        float e = __expf(p[i] - mn);
        ls += e;
        pexp[m * CH_ + i] = f2bf(e);
      }
      mrow[m] = mn; lrow[m] = ls; frow[m] = f;
    }
    __syncthreads();

    // ---- u update: rescale accumulators, then u_tile += exp(P) * x_chunk
    {
      float fr[8];
#pragma unroll
      for (int r = 0; r < 8; ++r) fr[r] = frow[r + mb];
      v16bf a = frag16(pexp + nn * CH_ + kb);     // A = exp(P) 16x32
#pragma unroll
      for (int t = 0; t < 6; ++t) {
#pragma unroll
        for (int r = 0; r < 8; ++r) acc[t][r] *= fr[r];
        v16bf bb = frag16(xT + (size_t)(j0 + t * 16 + nn) * CH_ + kb); // B = x^T 32x16
        acc[t] = __builtin_amdgcn_wmma_f32_16x16x32_bf16(false, a, false, bb,
                                                         (short)0, acc[t], false, false);
      }
    }
    __syncthreads();
  }

  // ---- write normalized context pre-projection: ubar = u / l
  float li[8];
#pragma unroll
  for (int r = 0; r < 8; ++r) li[r] = 1.0f / lrow[r + mb];
#pragma unroll
  for (int t = 0; t < 6; ++t) {
    int j = j0 + t * 16 + nn;
#pragma unroll
    for (int r = 0; r < 8; ++r)
      ubar[(size_t)b * 16 * H_ + (size_t)(r + mb) * H_ + j] = acc[t][r] * li[r];
  }
}

// ---------------- Kernel D: ctx = Wv ubar + bv ; cross = ctx @ out_w^T + out_b -
__global__ void __launch_bounds__(256)
ctx_cross_kernel(const float* __restrict__ ubar, const float* __restrict__ w,
                 const float* __restrict__ bias, const float* __restrict__ outw,
                 const float* __restrict__ outb, float* __restrict__ cross) {
  __shared__ float u[NH_ * H_];
  __shared__ float ctx[H_];
  int b = blockIdx.x;
  for (int i = threadIdx.x; i < NH_ * H_; i += blockDim.x) {
    int h = i / H_;
    u[i] = ubar[(size_t)b * 16 * H_ + (size_t)h * H_ + (i - h * H_)];
  }
  __syncthreads();
  for (int hd = threadIdx.x; hd < H_; hd += blockDim.x) {
    int h = hd / DH_;
    const float* wr = w + (size_t)(2 * H_ + hd) * H_;   // Wv row
    const float* ur = u + h * H_;
    float acc = bias[2 * H_ + hd];
    for (int j = 0; j < H_; ++j) acc += wr[j] * ur[j];
    ctx[hd] = acc;
  }
  __syncthreads();
  for (int i = threadIdx.x; i < H_; i += blockDim.x) {
    const float* wr = outw + (size_t)i * H_;
    float acc = outb[i];
    for (int j = 0; j < H_; ++j) acc += wr[j] * ctx[j];
    cross[b * H_ + i] = acc;
  }
}

// ---------------- Kernel E: fused vector + MLP heads ---------------------------
__global__ void __launch_bounds__(256)
head_kernel(const float* __restrict__ x, const float* __restrict__ cross,
            const float* __restrict__ aspq, const float* __restrict__ af,
            const float* __restrict__ lng, const float* __restrict__ lnb,
            const float* __restrict__ w1, const float* __restrict__ b1,
            const float* __restrict__ w2, const float* __restrict__ b2,
            const float* __restrict__ w3, const float* __restrict__ b3,
            const float* __restrict__ polw, const float* __restrict__ polb,
            const float* __restrict__ arw, const float* __restrict__ arb,
            float* __restrict__ out) {
  __shared__ float fused[FUSED_];
  __shared__ float h1[512];
  __shared__ float h2[128];
  int b = blockIdx.x;
  for (int j = threadIdx.x; j < H_; j += blockDim.x) {
    fused[j]          = x[(size_t)b * S_ * H_ + j];   // CLS token
    fused[H_ + j]     = cross[b * H_ + j];
    fused[2 * H_ + j] = aspq[b * H_ + j];
  }
  if (threadIdx.x == 0) {
    float mu = 0.f, m2 = 0.f;
    for (int i = 0; i < NF_; ++i) { float v = af[b * NF_ + i]; mu += v; m2 += v * v; }
    mu *= (1.0f / NF_); m2 = m2 * (1.0f / NF_) - mu * mu;
    float inv = rsqrtf(m2 + 1e-5f);
    for (int i = 0; i < NF_; ++i)
      fused[3 * H_ + i] = (af[b * NF_ + i] - mu) * inv * lng[i] + lnb[i];
  }
  __syncthreads();
  const float k = 0.70710678118654752f;
  for (int o = threadIdx.x; o < 512; o += blockDim.x) {
    const float* wr = w1 + (size_t)o * FUSED_;
    float acc = b1[o];
    for (int j = 0; j < FUSED_; ++j) acc += wr[j] * fused[j];
    h1[o] = 0.5f * acc * (1.0f + erff(acc * k));
  }
  __syncthreads();
  for (int o = threadIdx.x; o < 128; o += blockDim.x) {
    const float* wr = w2 + (size_t)o * 512;
    float acc = b2[o];
    for (int j = 0; j < 512; ++j) acc += wr[j] * h1[j];
    h2[o] = 0.5f * acc * (1.0f + erff(acc * k));
  }
  __syncthreads();
  if (threadIdx.x < 2) {
    int o = threadIdx.x;
    const float* wr = w3 + o * 128;
    float acc = b3[o];
    for (int j = 0; j < 128; ++j) acc += wr[j] * h2[j];
    out[b * 2 + o] = acc;                         // va
  } else if (threadIdx.x < 5) {
    int o = threadIdx.x - 2;
    const float* wr = polw + o * FUSED_;
    float acc = polb[o];
    for (int j = 0; j < FUSED_; ++j) acc += wr[j] * fused[j];
    out[2 * B_ + b * 3 + o] = acc;                // pol
  } else if (threadIdx.x < 8) {
    int o = threadIdx.x - 5;
    const float* wr = arw + o * FUSED_;
    float acc = arb[o];
    for (int j = 0; j < FUSED_; ++j) acc += wr[j] * fused[j];
    out[2 * B_ + 3 * B_ + b * 3 + o] = acc;       // aro
  }
}

extern "C" void kernel_launch(void* const* d_in, const int* in_sizes, int n_in,
                              void* d_out, int out_size, void* d_ws, size_t ws_size,
                              hipStream_t stream) {
  const float* x    = (const float*)d_in[0];
  const float* af   = (const float*)d_in[1];
  const int*   sep1 = (const int*)d_in[2];
  const int*   sep2 = (const int*)d_in[3];
  const float* ipw  = (const float*)d_in[4];
  const float* ipb  = (const float*)d_in[5];
  const float* outw = (const float*)d_in[6];
  const float* outb = (const float*)d_in[7];
  const float* lng  = (const float*)d_in[8];
  const float* lnb  = (const float*)d_in[9];
  const float* w1   = (const float*)d_in[10];
  const float* b1   = (const float*)d_in[11];
  const float* w2   = (const float*)d_in[12];
  const float* b2   = (const float*)d_in[13];
  const float* w3   = (const float*)d_in[14];
  const float* b3   = (const float*)d_in[15];
  const float* polw = (const float*)d_in[16];
  const float* polb = (const float*)d_in[17];
  const float* arw  = (const float*)d_in[18];
  const float* arb  = (const float*)d_in[19];

  char* ws = (char*)d_ws;
  float* aspq  = (float*)ws;                 ws += (size_t)B_ * H_ * 4;
  float* sbias = (float*)ws;                 ws += (size_t)B_ * 16 * 4;
  float* ubar  = (float*)ws;                 ws += (size_t)B_ * 16 * H_ * 4;
  float* cross = (float*)ws;                 ws += (size_t)B_ * H_ * 4;
  unsigned short* Tbf = (unsigned short*)ws; ws += (size_t)B_ * 16 * H_ * 2;

  asp_mean_kernel<<<B_, 256, 0, stream>>>(x, sep1, sep2, aspq);
  qT_kernel<<<B_, 256, 0, stream>>>(aspq, ipw, ipb, Tbf, sbias);

  // LDS: T(24576)+xrow(49152)+xT(49152)+pp(8192)+pexp(1024)+stats(256) = 132352 B
  const int smem = 16 * H_ * 2 + CH_ * H_ * 2 + H_ * CH_ * 2
                 + 4 * 16 * CH_ * 4 + 16 * CH_ * 2 + 4 * 16 * 4;
  (void)hipFuncSetAttribute(reinterpret_cast<const void*>(attn_kernel),
                            hipFuncAttributeMaxDynamicSharedMemorySize, smem);
  attn_kernel<<<B_, 256, smem, stream>>>(x, sep1, Tbf, sbias, ubar);

  ctx_cross_kernel<<<B_, 256, 0, stream>>>(ubar, ipw, ipb, outw, outb, cross);
  head_kernel<<<B_, 256, 0, stream>>>(x, cross, aspq, af, lng, lnb,
                                      w1, b1, w2, b2, w3, b3, polw, polb, arw, arb,
                                      (float*)d_out);
}
